// JEGOModule_57423712747667
// MI455X (gfx1250) — compile-verified, matrix operand order checked by reading, not run on hardware
//
#include <hip/hip_runtime.h>
#include <math.h>

// ---------------- problem constants ----------------
#define BATCH 4
#define CMOD  128      // D_MODEL
#define HH    64
#define WWID  64
#define LSEQ  2048
#define BL    (BATCH*LSEQ)   // 8192 tokens per sequence
#define DIN   256            // D_INNER
#define DTR   8
#define DST   16
#define PADW  66             // 64 + 2 halo
#define KCONV 1152           // 9 taps * 128 cin

// ---------------- WMMA / vector types ----------------
typedef __bf16 bf16_t;
typedef __attribute__((ext_vector_type(16))) __bf16 bfrag16;
typedef __attribute__((ext_vector_type(8)))  float  f32x8;
typedef __attribute__((ext_vector_type(4)))  unsigned int u32x4;
typedef __attribute__((ext_vector_type(8)))  int i32x8;
typedef __attribute__((ext_vector_type(4)))  int i32x4;

__device__ __forceinline__ f32x8 wmma_bf16(bfrag16 a, bfrag16 b, f32x8 c) {
    return __builtin_amdgcn_wmma_f32_16x16x32_bf16(
        /*neg_a=*/false, a, /*neg_b=*/false, b,
        /*c_mod=*/(short)0, c, /*reuse_a=*/false, /*reuse_b=*/false);
}

// =====================================================================
// 0) one-time (per launch) weight conversion to bf16
// =====================================================================
__global__ void k_cvt_f32_bf16(const float* __restrict__ src, bf16_t* __restrict__ dst, int n) {
    int i = blockIdx.x * blockDim.x + threadIdx.x;
    if (i < n) dst[i] = (bf16_t)src[i];
}

// agg conv weights (cout,cin,3,3) -> bf16 [cout][t*128+cin]  (K-contiguous rows)
__global__ void k_cvt_convw(const float* __restrict__ w, bf16_t* __restrict__ out) {
    int i = blockIdx.x * blockDim.x + threadIdx.x;   // cout*1152 + t*128 + cin
    int cout = i / KCONV, r = i % KCONV, t = r >> 7, cin = r & 127;
    out[i] = (bf16_t)w[((size_t)cout * CMOD + cin) * 9 + t];
}

// =====================================================================
// 1) Direction gather + RMSNorm (fp32 residual copy + bf16 normalized)
// =====================================================================
__device__ __forceinline__ float gather_val(const float* fi, const float* fj,
                                            int k, int b, int l, int c) {
    int m = (k >= 2) ? (LSEQ - 1 - l) : l;  // seqs 2,3 are reversed
    int i = m >> 6;
    int j = m & 63;
    int row, col;
    if      (k == 0) { row = 2*i;     col = 2*j;     }
    else if (k == 2) { row = 2*i;     col = 2*j + 1; }
    else if (k == 1) { row = 2*j + 1; col = 2*i + 1; }
    else             { row = 2*j + 1; col = 2*i;     }
    const float* src; int h, w;
    if (k == 0 || k == 2) {            // horizontal concat (width 128)
        if (col < WWID) { src = fi; h = row; w = col; }
        else            { src = fj; h = row; w = col - WWID; }
    } else {                           // vertical concat (height 128)
        if (row < HH)   { src = fi; h = row; w = col; }
        else            { src = fj; h = row - HH; w = col; }
    }
    return src[(((size_t)b*CMOD + c)*HH + h)*WWID + w];
}

__global__ void k_gather_norm(const float* __restrict__ fi, const float* __restrict__ fj,
                              const float* __restrict__ nw, int k,
                              float* __restrict__ xseq, bf16_t* __restrict__ hbf) {
    int tok = blockIdx.x;
    int c   = threadIdx.x;
    int b = tok >> 11, l = tok & 2047;
    float v = gather_val(fi, fj, k, b, l, c);
    __shared__ float red[CMOD];
    red[c] = v * v;
    __syncthreads();
    for (int s = 64; s > 0; s >>= 1) {
        if (c < s) red[c] += red[c + s];
        __syncthreads();
    }
    float sc = rsqrtf(red[0] * (1.0f / CMOD) + 1e-5f);
    size_t idx = (size_t)tok * CMOD + c;
    xseq[idx] = v;
    hbf[idx]  = (bf16_t)(v * sc * nw[c]);
}

// =====================================================================
// 2) Generic WMMA GEMM:  C[M,N] = A[M,K](bf16) * W[N,K]^T(bf16) (+resid)
// =====================================================================
__global__ void k_gemm_bf16(const bf16_t* __restrict__ A, const bf16_t* __restrict__ Wnk,
                            const float* __restrict__ resid, float* __restrict__ Cout,
                            int M, int N, int K) {
    int lane = threadIdx.x & 31;
    int wv   = threadIdx.x >> 5;
    int tile = blockIdx.x * 4 + wv;
    int ntiles = N >> 4;
    int mt = tile / ntiles, nt = tile % ntiles;
    int arow = mt * 16 + (lane & 15);
    int ncol = nt * 16 + (lane & 15);
    int koff = (lane >> 4) << 3;
    f32x8 acc = {0.f,0.f,0.f,0.f,0.f,0.f,0.f,0.f};
    for (int k0 = 0; k0 < K; k0 += 32) {
        bfrag16 a, bw;
        const bf16_t* pa = A   + (size_t)arow * K + k0 + koff;
        const bf16_t* pb = Wnk + (size_t)ncol * K + k0 + koff;
        __builtin_prefetch(pa + 32, 0, 1);   // global_prefetch_b8, next K chunk
#pragma unroll
        for (int e = 0; e < 8; ++e) {
            a[e]      = pa[e];
            a[8 + e]  = pa[16 + e];
            bw[e]     = pb[e];
            bw[8 + e] = pb[16 + e];
        }
        acc = wmma_bf16(a, bw, acc);
    }
    int rbase = mt * 16 + ((lane >> 4) << 3);
    int col   = nt * 16 + (lane & 15);
#pragma unroll
    for (int r = 0; r < 8; ++r) {
        size_t idx = (size_t)(rbase + r) * N + col;
        float v = acc[r];
        if (resid) v += resid[idx];
        Cout[idx] = v;
    }
}

// =====================================================================
// 3) causal depth-4 conv1d over L + SiLU
// =====================================================================
__global__ void k_conv_silu(const float* __restrict__ xz, const float* __restrict__ cw,
                            const float* __restrict__ cb, float* __restrict__ xcf) {
    int idx = blockIdx.x * blockDim.x + threadIdx.x;   // t*256 + e
    int e = idx & 255, t = idx >> 8, l = t & 2047;
    float s = cb[e];
#pragma unroll
    for (int kk = 0; kk < 4; ++kk) {
        int lm = l + kk - 3;
        if (lm >= 0) s += cw[e * 4 + kk] * xz[(size_t)(t + kk - 3) * 512 + e];
    }
    float r = s / (1.f + __expf(-s));
    xcf[idx] = r;
}

// =====================================================================
// 4) x_proj (N=40) and delta = softplus(dt @ dtw^T + dtb)
// =====================================================================
__global__ void k_xproj(const float* __restrict__ xcf, const float* __restrict__ xpw,
                        float* __restrict__ dbc) {
    int idx = blockIdx.x * blockDim.x + threadIdx.x;   // t*40 + n
    int n = idx % 40, t = idx / 40;
    const float* a = xcf + (size_t)t * DIN;
    const float* w = xpw + (size_t)n * DIN;
    float s = 0.f;
    for (int kk = 0; kk < DIN; ++kk) s = fmaf(a[kk], w[kk], s);
    dbc[idx] = s;
}

__global__ void k_delta(const float* __restrict__ dbc, const float* __restrict__ dtw,
                        const float* __restrict__ dtb, float* __restrict__ delta) {
    int idx = blockIdx.x * blockDim.x + threadIdx.x;   // t*256 + d
    int d = idx & 255, t = idx >> 8;
    float s = dtb[d];
    const float* r = dbc + (size_t)t * 40;
#pragma unroll
    for (int q = 0; q < DTR; ++q) s = fmaf(r[q], dtw[d * DTR + q], s);
    delta[idx] = (s > 20.f) ? s : log1pf(__expf(s));
}

// =====================================================================
// 5) selective scan (16 lanes = 16 states), fused D-skip + silu(z) gate
// =====================================================================
__global__ void k_scan(const float* __restrict__ delta, const float* __restrict__ xcf,
                       const float* __restrict__ dbc, const float* __restrict__ xz,
                       const float* __restrict__ alog, const float* __restrict__ dpar,
                       bf16_t* __restrict__ ybf) {
    int lane = threadIdx.x & 31;
    int gwav = (blockIdx.x * blockDim.x + threadIdx.x) >> 5;  // 0..511
    int half = lane >> 4, n = lane & 15;
    int pair = gwav * 2 + half;                // (b,d)
    int b = pair >> 8, d = pair & 255;
    float An = -__expf(alog[d * DST + n]);
    float Dd = dpar[d];
    float h = 0.f;
    for (int l = 0; l < LSEQ; ++l) {
        size_t t = (size_t)b * LSEQ + l;
        float dl = delta[t * DIN + d];
        float u  = xcf[t * DIN + d];
        const float* bc = dbc + t * 40;
        float dA = __expf(dl * An);
        h = fmaf(dA, h, dl * u * bc[8 + n]);
        float acc = h * bc[24 + n];
        acc += __shfl_xor(acc, 8, 16);
        acc += __shfl_xor(acc, 4, 16);
        acc += __shfl_xor(acc, 2, 16);
        acc += __shfl_xor(acc, 1, 16);
        if (n == 0) {
            float y = acc + u * Dd;
            float z = xz[t * 512 + DIN + d];
            float g = y * (z / (1.f + __expf(-z)));
            ybf[t * DIN + d] = (bf16_t)g;
        }
    }
}

// =====================================================================
// 6) inverse-scan scatter -> zero-padded NHWC bf16 feature (8,66,66,128)
// =====================================================================
__global__ void k_scatter(const float* __restrict__ outall, bf16_t* __restrict__ featp) {
    int pix = blockIdx.x;
    int c = threadIdx.x;
    int xx = pix % PADW, tmp = pix / PADW, yy = tmp % PADW, nn = tmp / PADW;
    size_t dst = (size_t)pix * CMOD + c;
    if (xx == 0 || xx == PADW-1 || yy == 0 || yy == PADW-1) { featp[dst] = (bf16_t)0.f; return; }
    int x = xx - 1, y = yy - 1;
    int b = nn & 3, jofs = (nn >= 4) ? 32 : 0;
    int k, i, j, l;
    if ((y & 1) == 0) {
        if ((x & 1) == 0) { k = 0; i = y >> 1;     j = (x >> 1)     + jofs; l = i*64 + j; }
        else              { k = 2; i = y >> 1;     j = ((x-1) >> 1) + jofs; l = 2047 - (i*64 + j); }
    } else {
        if (x & 1)        { k = 1; i = (x-1) >> 1; j = ((y-1) >> 1) + jofs; l = i*64 + j; }
        else              { k = 3; i = x >> 1;     j = ((y-1) >> 1) + jofs; l = 2047 - (i*64 + j); }
    }
    featp[dst] = (bf16_t)outall[(((size_t)k*BATCH + b)*LSEQ + l)*CMOD + c];
}

__global__ void k_fill_bf16_zero(bf16_t* __restrict__ p, size_t nelem) {
    size_t i = (size_t)blockIdx.x * blockDim.x + threadIdx.x;
    if (i < nelem) p[i] = (bf16_t)0.f;
}

// =====================================================================
// 7a) gated 3x3 conv: g = gelu(conv_W)*conv_V, shared A frag, bf16 weights
// =====================================================================
__global__ void k_conv_gated(const bf16_t* __restrict__ featp, const bf16_t* __restrict__ WgB,
                             const bf16_t* __restrict__ WvB, bf16_t* __restrict__ gout) {
    int lane = threadIdx.x & 31;
    int wv   = threadIdx.x >> 5;
    int tile = blockIdx.x * 4 + wv;            // 2048 ptiles * 8 ntiles
    int pt = tile >> 3, nt = tile & 7;
    int prow = pt * 16 + (lane & 15);
    int nn = prow >> 12, y = (prow >> 6) & 63, xa = prow & 63;
    int koff = (lane >> 4) << 3;
    int ncol = nt * 16 + (lane & 15);
    f32x8 accw = {0.f,0.f,0.f,0.f,0.f,0.f,0.f,0.f};
    f32x8 accv = {0.f,0.f,0.f,0.f,0.f,0.f,0.f,0.f};
    for (int t = 0; t < 9; ++t) {
        int dy = t / 3, dx = t % 3;
        const bf16_t* base = featp + (((size_t)nn*PADW + y + dy)*PADW + (xa + dx)) * CMOD;
        for (int cc = 0; cc < CMOD; cc += 32) {
            bfrag16 a, bw, bv;
            const bf16_t* pa = base + cc + koff;
            const bf16_t* pw = WgB + (size_t)ncol*KCONV + t*CMOD + cc + koff;
            const bf16_t* pv = WvB + (size_t)ncol*KCONV + t*CMOD + cc + koff;
#pragma unroll
            for (int e = 0; e < 8; ++e) {
                a[e]  = pa[e];  a[8+e]  = pa[16+e];
                bw[e] = pw[e];  bw[8+e] = pw[16+e];
                bv[e] = pv[e];  bv[8+e] = pv[16+e];
            }
            accw = wmma_bf16(a, bw, accw);
            accv = wmma_bf16(a, bv, accv);
        }
    }
    int px0 = (pt * 16) & 63;
    int ny  = (pt * 16) >> 6;
    int n_s = ny >> 6, y_s = ny & 63;
#pragma unroll
    for (int r = 0; r < 8; ++r) {
        int x = px0 + ((lane >> 4) << 3) + r;
        float gw = accw[r];
        float g  = 0.5f * gw * (1.f + erff(gw * 0.70710678118f)) * accv[r];
        gout[(((size_t)n_s*PADW + y_s + 1)*PADW + (x + 1))*CMOD + ncol] = (bf16_t)g;
    }
}

// =====================================================================
// 7b) final 3x3 conv: weights staged to LDS via Tensor Data Mover.
//     4 waves/block share one 16-cout weight slice (16 x 1152 bf16 = 36KB)
// =====================================================================
__global__ void k_conv_final(const bf16_t* __restrict__ gp, const bf16_t* __restrict__ W2b,
                             float* __restrict__ out) {
    __shared__ bf16_t smW[16 * KCONV];         // single LDS alloc -> base offset 0
    int lane = threadIdx.x & 31;
    int wv   = threadIdx.x >> 5;
    int tile = blockIdx.x * 4 + wv;            // nt-major: all 4 waves share nt
    int nt = tile >> 11, pt = tile & 2047;
    int ntbase = nt * 16;

    if (wv == 0) {
        // TDM: 2-D tile load, 16 rows (couts) x 1152 elements, 2B each
        unsigned long long ga = (unsigned long long)(size_t)(W2b + (size_t)ntbase * KCONV);
        u32x4 g0;
        g0[0] = 1u;                                          // count=1, user mode
        g0[1] = 0u;                                          // lds_addr = 0 (smW)
        g0[2] = (unsigned int)(ga & 0xFFFFFFFFu);            // global_addr[31:0]
        g0[3] = (unsigned int)((ga >> 32) & 0x1FFFFFFu) | (2u << 30);  // addr[56:32] | type=2
        i32x8 g1;
        g1[0] = (int)(1u << 16);                             // data_size=1 (2 bytes)
        g1[1] = (int)((KCONV & 0xFFFFu) << 16);              // tensor_dim0 low16 @bit48
        g1[2] = (int)((KCONV >> 16) | ((128u & 0xFFFFu) << 16)); // dim0 hi | tensor_dim1 lo
        g1[3] = (int)(KCONV << 16);                          // tensor_dim1 hi=0 | tile_dim0
        g1[4] = 16;                                          // tile_dim1=16, tile_dim2=0
        g1[5] = KCONV;                                       // tensor_dim0_stride[31:0]
        g1[6] = 0;                                           // stride0 hi | stride1 lo
        g1[7] = 0;
        i32x4 zz4 = {0, 0, 0, 0};
        i32x8 zz8 = {0, 0, 0, 0, 0, 0, 0, 0};
        __builtin_amdgcn_tensor_load_to_lds(g0, g1, zz4, zz4, zz8, 0);
        __builtin_amdgcn_s_wait_tensorcnt((short)0);
    }
    __syncthreads();

    int prow = pt * 16 + (lane & 15);
    int nn = prow >> 12, y = (prow >> 6) & 63, xa = prow & 63;
    int koff = (lane >> 4) << 3;
    int nloc = lane & 15;
    f32x8 acc = {0.f,0.f,0.f,0.f,0.f,0.f,0.f,0.f};
    for (int t = 0; t < 9; ++t) {
        int dy = t / 3, dx = t % 3;
        const bf16_t* base = gp + (((size_t)nn*PADW + y + dy)*PADW + (xa + dx)) * CMOD;
        for (int cc = 0; cc < CMOD; cc += 32) {
            bfrag16 a, bw;
            const bf16_t* pa = base + cc + koff;
            const bf16_t* pb = smW + nloc*KCONV + t*CMOD + cc + koff;   // LDS reads
#pragma unroll
            for (int e = 0; e < 8; ++e) {
                a[e]  = pa[e];  a[8+e]  = pa[16+e];
                bw[e] = pb[e];  bw[8+e] = pb[16+e];
            }
            acc = wmma_bf16(a, bw, acc);
        }
    }
    int px0 = (pt * 16) & 63;
    int ny  = (pt * 16) >> 6;
    int n_s = ny >> 6, y_s = ny & 63;
    int ncol = ntbase + nloc;
#pragma unroll
    for (int r = 0; r < 8; ++r) {
        int x = px0 + ((lane >> 4) << 3) + r;
        out[(((size_t)n_s*CMOD + ncol)*HH + y_s)*WWID + x] = acc[r];
    }
}

// =====================================================================
// host side
// =====================================================================
extern "C" void kernel_launch(void* const* d_in, const int* in_sizes, int n_in,
                              void* d_out, int out_size, void* d_ws, size_t ws_size,
                              hipStream_t stream) {
    (void)in_sizes; (void)n_in; (void)out_size; (void)ws_size;
    const float* fi        = (const float*)d_in[0];
    const float* fj        = (const float*)d_in[1];
    const float* norm_w    = (const float*)d_in[2];
    const float* in_proj_w = (const float*)d_in[3];
    const float* conv_w    = (const float*)d_in[4];
    const float* conv_b    = (const float*)d_in[5];
    const float* x_proj_w  = (const float*)d_in[6];
    const float* dt_w      = (const float*)d_in[7];
    const float* dt_b      = (const float*)d_in[8];
    const float* A_log     = (const float*)d_in[9];
    const float* Dpar      = (const float*)d_in[10];
    const float* out_proj_w= (const float*)d_in[11];
    const float* agg_W     = (const float*)d_in[12];
    const float* agg_V     = (const float*)d_in[13];
    const float* agg_W2    = (const float*)d_in[14];

    char* ws = (char*)d_ws;
    size_t off = 0;
    auto take = [&](size_t bytes) -> char* {
        char* p = ws + off;
        off = (off + bytes + 255) & ~(size_t)255;
        return p;
    };
    float*  XSEQ   = (float*) take((size_t)BL * CMOD * 4);
    bf16_t* HBF    = (bf16_t*)take((size_t)BL * CMOD * 2);
    float*  XZ     = (float*) take((size_t)BL * 512 * 4);
    float*  XCF    = (float*) take((size_t)BL * DIN * 4);
    float*  DBC    = (float*) take((size_t)BL * 40 * 4);
    float*  DELTA  = (float*) take((size_t)BL * DIN * 4);
    bf16_t* YBF    = (bf16_t*)take((size_t)BL * DIN * 2);
    float*  OUTALL = (float*) take((size_t)4 * BL * CMOD * 4);
    bf16_t* FEATP  = (bf16_t*)take((size_t)8 * PADW * PADW * CMOD * 2);
    bf16_t* GP     = (bf16_t*)take((size_t)8 * PADW * PADW * CMOD * 2);
    bf16_t* WIPB   = (bf16_t*)take((size_t)4 * 512 * CMOD * 2);
    bf16_t* WOPB   = (bf16_t*)take((size_t)4 * CMOD * DIN * 2);
    bf16_t* WAGW   = (bf16_t*)take((size_t)CMOD * KCONV * 2);
    bf16_t* WAGV   = (bf16_t*)take((size_t)CMOD * KCONV * 2);
    bf16_t* WAG2   = (bf16_t*)take((size_t)CMOD * KCONV * 2);

    // weight preconversion (bf16 + conv re-layout)
    k_cvt_f32_bf16<<<(4*512*CMOD)/256, 256, 0, stream>>>(in_proj_w, WIPB, 4*512*CMOD);
    k_cvt_f32_bf16<<<(4*CMOD*DIN)/256, 256, 0, stream>>>(out_proj_w, WOPB, 4*CMOD*DIN);
    k_cvt_convw<<<(CMOD*KCONV)/256, 256, 0, stream>>>(agg_W, WAGW);
    k_cvt_convw<<<(CMOD*KCONV)/256, 256, 0, stream>>>(agg_V, WAGV);
    k_cvt_convw<<<(CMOD*KCONV)/256, 256, 0, stream>>>(agg_W2, WAG2);

    for (int k = 0; k < 4; ++k) {
        k_gather_norm<<<BL, CMOD, 0, stream>>>(fi, fj, norm_w + k*CMOD, k, XSEQ, HBF);
        k_gemm_bf16<<<(512*32)/4, 128, 0, stream>>>(HBF, WIPB + (size_t)k*512*CMOD,
                                                    nullptr, XZ, BL, 512, CMOD);
        k_conv_silu<<<(BL*DIN)/256, 256, 0, stream>>>(XZ, conv_w + k*DIN*4, conv_b + k*DIN, XCF);
        k_xproj<<<(BL*40)/256, 256, 0, stream>>>(XCF, x_proj_w + (size_t)k*40*DIN, DBC);
        k_delta<<<(BL*DIN)/256, 256, 0, stream>>>(DBC, dt_w + k*DIN*DTR, dt_b + k*DIN, DELTA);
        k_scan<<<128, 128, 0, stream>>>(DELTA, XCF, DBC, XZ,
                                        A_log + (size_t)k*DIN*DST, Dpar + k*DIN, YBF);
        k_gemm_bf16<<<(512*8)/4, 128, 0, stream>>>(YBF, WOPB + (size_t)k*CMOD*DIN,
                                                   XSEQ, OUTALL + (size_t)k*BL*CMOD,
                                                   BL, CMOD, DIN);
    }
    k_scatter<<<8*PADW*PADW, CMOD, 0, stream>>>(OUTALL, FEATP);
    size_t gn = (size_t)8 * PADW * PADW * CMOD;
    k_fill_bf16_zero<<<(int)((gn + 255) / 256), 256, 0, stream>>>(GP, gn);
    k_conv_gated<<<(2048*8)/4, 128, 0, stream>>>(FEATP, WAGW, WAGV, GP);
    k_conv_final<<<(2048*8)/4, 128, 0, stream>>>(GP, WAG2, (float*)d_out);
}
